// LstmRNN_68143951118863
// MI455X (gfx1250) — matrix operand order repaired and checked by reading, not compiled, not run on hardware
//
#include <hip/hip_runtime.h>

typedef _Float16 v16h __attribute__((ext_vector_type(16)));
typedef _Float16 v8h  __attribute__((ext_vector_type(8)));
typedef float    v8f  __attribute__((ext_vector_type(8)));

#define SEQ   2048
#define BATCH 64
#define HID   128
#define G4H   512          // 4*HID (gate dim)
#define MROWS (SEQ * BATCH)

__device__ __forceinline__ float sigmf(float x) {
    return 1.0f / (1.0f + __expf(-x));
}

// ---------------------------------------------------------------------------
// Input-projection GEMM: gx[sb][n] = sum_k in[sb][k] * w_ih[n][k] + b[n]
// One 16x16 output tile per wave; fp32 operands converted to fp16 on the fly,
// fp32 accumulation via v_wmma_f32_16x16x32_f16. blockIdx.y selects direction.
// ---------------------------------------------------------------------------
__global__ void gx_gemm_kernel(const float* __restrict__ in, int Kin,
                               const float* __restrict__ wF, const float* __restrict__ bF,
                               float* __restrict__ gxF,
                               const float* __restrict__ wR, const float* __restrict__ bR,
                               float* __restrict__ gxR)
{
    const float* w  = blockIdx.y ? wR : wF;
    const float* bb = blockIdx.y ? bR : bF;
    float*       gx = blockIdx.y ? gxR : gxF;

    int wave = threadIdx.x >> 5;
    int lane = threadIdx.x & 31;
    int ln   = lane & 15;
    int hi   = lane >> 4;                 // 0 or 1 (half-wave)

    int tile  = blockIdx.x * 8 + wave;    // 8 waves per block, 1 tile per wave
    int mTile = tile >> 5;                // 32 N-tiles (512/16)
    int nTile = tile & 31;

    int n    = nTile * 16 + ln;           // output column (0..511)
    int arow = mTile * 16 + ln;           // A row handled by this lane

    v8f acc;
    float bv = bb[n];
    #pragma unroll
    for (int r = 0; r < 8; ++r) acc[r] = bv;

    int nk = Kin >> 5;                    // K tiles of 32
    for (int kt = 0; kt < nk; ++kt) {
        // A fragment (rows of `in`): K runs [k0,k0+7] and [k0+16,k0+23]
        const float* ap = in + (size_t)arow * Kin + kt * 32 + 8 * hi;
        v16h a;
        #pragma unroll
        for (int j = 0; j < 8; ++j) {
            a[j]     = (_Float16)ap[j];
            a[j + 8] = (_Float16)ap[16 + j];
        }
        // B fragment: B[k][n] = w_ih[n][k]; K runs [k0,k0+7] and [k0+16,k0+23]
        const float* bp = w + (size_t)n * Kin + kt * 32 + 8 * hi;
        v16h b;
        #pragma unroll
        for (int j = 0; j < 8; ++j) { b[j] = (_Float16)bp[j]; b[j + 8] = (_Float16)bp[16 + j]; }

        acc = __builtin_amdgcn_wmma_f32_16x16x32_f16(false, a, false, b,
                                                     (short)0, acc, false, false);
    }

    #pragma unroll
    for (int r = 0; r < 8; ++r) {
        int row = mTile * 16 + r + 8 * hi;
        gx[(size_t)row * G4H + n] = acc[r];
    }
}

// ---------------------------------------------------------------------------
// Persistent recurrent scan: one workgroup (8 waves) per direction.
// g_t = gx[t] + h @ w_hh^T ; gates i,f,g,o ; c,h update.
// w_hh held fp16 in VGPRs (weights stationary), h ping-pongs in LDS fp16,
// c stays in VGPRs for the entire sequence.
// Wave w owns output columns [w*16, w*16+16) of each gate block.
// ---------------------------------------------------------------------------
__global__ void lstm_scan_kernel(const float* __restrict__ gxF, const float* __restrict__ gxR,
                                 const float* __restrict__ whhF, const float* __restrict__ whhR,
                                 float* __restrict__ out)
{
    int dir = blockIdx.x;
    const float* gx  = dir ? gxR : gxF;
    const float* whh = dir ? whhR : whhF;
    int dirOff = dir * HID;

    __shared__ __align__(16) _Float16 hbuf[2][BATCH * HID];  // 2 x 16 KB ping-pong

    int tid  = threadIdx.x;
    int wv   = tid >> 5;
    int lane = tid & 31;
    int ln   = lane & 15;
    int hi   = lane >> 4;

    for (int i = tid; i < BATCH * HID; i += blockDim.x) hbuf[0][i] = (_Float16)0.f;
    __syncthreads();

    // Resident B fragments: B[k][n] = whh[n][k], n = gt*128 + wv*16 + ln
    v16h Bf[4][4];
    #pragma unroll
    for (int kt = 0; kt < 4; ++kt) {
        #pragma unroll
        for (int gt = 0; gt < 4; ++gt) {
            int nn = gt * HID + wv * 16 + ln;
            const float* bp = whh + (size_t)nn * HID + kt * 32 + 8 * hi;
            #pragma unroll
            for (int j = 0; j < 8; ++j) { Bf[kt][gt][j] = (_Float16)bp[j]; Bf[kt][gt][j + 8] = (_Float16)bp[16 + j]; }
        }
    }

    // Cell state registers: tile (m, wv), accumulator layout
    v8f c[4];
    #pragma unroll
    for (int m = 0; m < 4; ++m) {
        #pragma unroll
        for (int r = 0; r < 8; ++r) c[m][r] = 0.f;
    }

    for (int step = 0; step < SEQ; ++step) {
        int t = dir ? (SEQ - 1 - step) : step;
        const float* gxt = gx + (size_t)t * BATCH * G4H;
        int cur = step & 1, nxt = cur ^ 1;

        // Prefetch next step's gx slab (64*512 floats) across the block
        if (step + 1 < SEQ) {
            int tn = dir ? (SEQ - 2 - step) : (step + 1);
            const float* pf = gx + (size_t)tn * BATCH * G4H;
            __builtin_prefetch(pf + (size_t)tid * 128, 0, 0);
            __builtin_prefetch(pf + (size_t)tid * 128 + 64, 0, 0);
        }

        #pragma unroll
        for (int m = 0; m < 4; ++m) {
            // Init accumulators with precomputed input gates (gx already has +b)
            v8f acc[4];
            #pragma unroll
            for (int gt = 0; gt < 4; ++gt) {
                #pragma unroll
                for (int r = 0; r < 8; ++r)
                    acc[gt][r] = gxt[(size_t)(m * 16 + r + 8 * hi) * G4H
                                     + gt * HID + wv * 16 + ln];
            }

            // h @ w_hh^T over K=128 (4 k-tiles of 32)
            #pragma unroll
            for (int kt = 0; kt < 4; ++kt) {
                const _Float16* hp = &hbuf[cur][(m * 16 + ln) * HID + kt * 32 + 8 * hi];
                v8h alo = *(const v8h*)(hp);
                v8h ahi = *(const v8h*)(hp + 16);
                v16h a;
                #pragma unroll
                for (int j = 0; j < 8; ++j) { a[j] = alo[j]; a[j + 8] = ahi[j]; }
                #pragma unroll
                for (int gt = 0; gt < 4; ++gt)
                    acc[gt] = __builtin_amdgcn_wmma_f32_16x16x32_f16(
                        false, a, false, Bf[kt][gt], (short)0, acc[gt], false, false);
            }

            // Gate math + state update (PyTorch gate order: i, f, g, o)
            #pragma unroll
            for (int r = 0; r < 8; ++r) {
                float ig = sigmf(acc[0][r]);
                float fg = sigmf(acc[1][r]);
                float gg = tanhf(acc[2][r]);
                float og = sigmf(acc[3][r]);
                float cn = fg * c[m][r] + ig * gg;
                c[m][r] = cn;
                float hv = og * tanhf(cn);
                int row = m * 16 + r + 8 * hi;   // batch index
                int col = wv * 16 + ln;          // hidden unit
                hbuf[nxt][row * HID + col] = (_Float16)hv;
                out[((size_t)t * BATCH + row) * (2 * HID) + dirOff + col] = hv;
            }
        }
        __syncthreads();
    }
}

// ---------------------------------------------------------------------------
// Final linear: y[sb] = out1[sb][:] . w_out + b_out   (O = 1)
// ---------------------------------------------------------------------------
__global__ void out_linear_kernel(const float* __restrict__ h2,
                                  const float* __restrict__ wout,
                                  const float* __restrict__ bout,
                                  float* __restrict__ y)
{
    int idx = blockIdx.x * blockDim.x + threadIdx.x;   // 0 .. 131071
    const float* row = h2 + (size_t)idx * 256;
    float acc = 0.f;
    #pragma unroll 8
    for (int k = 0; k < 256; ++k) acc += row[k] * wout[k];
    y[idx] = acc + bout[0];
}

// ---------------------------------------------------------------------------
extern "C" void kernel_launch(void* const* d_in, const int* in_sizes, int n_in,
                              void* d_out, int out_size, void* d_ws, size_t ws_size,
                              hipStream_t stream)
{
    const float* x       = (const float*)d_in[0];
    const float* w_ih_f0 = (const float*)d_in[1];
    const float* w_hh_f0 = (const float*)d_in[2];
    const float* b_f0    = (const float*)d_in[3];
    const float* w_ih_r0 = (const float*)d_in[4];
    const float* w_hh_r0 = (const float*)d_in[5];
    const float* b_r0    = (const float*)d_in[6];
    const float* w_ih_f1 = (const float*)d_in[7];
    const float* w_hh_f1 = (const float*)d_in[8];
    const float* b_f1    = (const float*)d_in[9];
    const float* w_ih_r1 = (const float*)d_in[10];
    const float* w_hh_r1 = (const float*)d_in[11];
    const float* b_r1    = (const float*)d_in[12];
    const float* w_out   = (const float*)d_in[13];
    const float* b_out   = (const float*)d_in[14];
    float* y             = (float*)d_out;

    // Workspace layout (floats): gxF | gxR | out0 | out1
    const size_t GXN  = (size_t)MROWS * G4H;       // 67,108,864
    const size_t OUTN = (size_t)MROWS * 2 * HID;   // 33,554,432
    float* ws   = (float*)d_ws;
    float* gxF  = ws;
    float* gxR  = ws + GXN;
    float* out0 = ws + 2 * GXN;
    float* out1 = out0 + OUTN;

    // (MROWS/16)*(512/16) tiles, 8 tiles (waves) per 256-thread block
    const int gemmBlocks = (MROWS / 16) * (G4H / 16) / 8;   // 32768

    // Layer 0
    gx_gemm_kernel<<<dim3(gemmBlocks, 2), 256, 0, stream>>>(
        x, 64, w_ih_f0, b_f0, gxF, w_ih_r0, b_r0, gxR);
    lstm_scan_kernel<<<2, 256, 0, stream>>>(gxF, gxR, w_hh_f0, w_hh_r0, out0);

    // Layer 1 (input = out0, Kin = 256); gx buffers reused
    gx_gemm_kernel<<<dim3(gemmBlocks, 2), 256, 0, stream>>>(
        out0, 256, w_ih_f1, b_f1, gxF, w_ih_r1, b_r1, gxR);
    lstm_scan_kernel<<<2, 256, 0, stream>>>(gxF, gxR, w_hh_f1, w_hh_r1, out1);

    // Final projection
    out_linear_kernel<<<MROWS / 256, 256, 0, stream>>>(out1, w_out, b_out, y);
}
